// SingleAttentionHead_54150947668668
// MI455X (gfx1250) — matrix operand order repaired and checked by reading, not compiled, not run on hardware
//
#include <hip/hip_runtime.h>
#include <cstdint>

#define EDIM 1024
#define HDIM 64
#define BDIM 4
#define SDIM 4096

typedef __attribute__((ext_vector_type(16))) __bf16 v16bf;
typedef __attribute__((ext_vector_type(8)))  float  v8f;
typedef __attribute__((ext_vector_type(4)))  unsigned int u32x4;
typedef __attribute__((ext_vector_type(8)))  unsigned int u32x8;

union U32B { uint4 u[2]; v16bf v; };

static __device__ inline v16bf ldg_v16bf(const __bf16* p) {
    U32B t;
    const uint4* q = (const uint4*)p;
    t.u[0] = q[0]; t.u[1] = q[1];
    return t.v;
}
static __device__ inline v16bf ld2_v16bf(const __bf16* p0, const __bf16* p1) {
    U32B t;
    t.u[0] = *(const uint4*)p0;
    t.u[1] = *(const uint4*)p1;
    return t.v;
}

// Issue a 2D TDM load: global (row-major, 2-byte elems) -> LDS.
// g0: count=1 | lds byte addr | 57-bit global byte addr | type=2.
// g1: dims/strides, prebuilt constant. Groups 2/3 omitted (NULL, 2D tensor).
// lds_ptr is passed into the asm (unreferenced) so the destination buffer
// ESCAPES: with the "memory" clobber the compiler must treat the asm as a
// writer of that LDS object and keep/reload subsequent ds loads.
static __device__ inline void tdm_load_2d(void* lds_ptr, unsigned lds_byte,
                                          unsigned long long gaddr,
                                          u32x8 g1) {
    u32x4 g0;
    g0[0] = 1u;                                   // count=1, user mode
    g0[1] = (unsigned)__builtin_amdgcn_readfirstlane((int)lds_byte);
    g0[2] = (unsigned)__builtin_amdgcn_readfirstlane((int)(unsigned)(gaddr & 0xffffffffull));
    g0[3] = ((unsigned)__builtin_amdgcn_readfirstlane((int)(unsigned)(gaddr >> 32)) & 0x01ffffffu)
            | 0x80000000u;                        // type=2 ("image")
    asm volatile("tensor_load_to_lds %0, %1"
                 :: "s"(g0), "s"(g1), "v"(lds_ptr)
                 : "memory");
}

// ---------------- Kernel 0: pack weights f32 -> bf16 in WMMA B-tile layout --
// packed[mat][((kt*4 + nt)*32 + lane)*16 + j] = W[(kt*32 + K)*H + nt*16 + n]
//   with n = lane&15, K = (lane>>4)*16 + j   (contiguous k order, same as the A tiles)
__global__ void pack_w_kernel(const float* __restrict__ Wq,
                              const float* __restrict__ Wk,
                              const float* __restrict__ Wv,
                              __bf16* __restrict__ out) {
    int idx = blockIdx.x * blockDim.x + threadIdx.x;
    if (idx >= 3 * EDIM * HDIM) return;
    int mat = idx / (EDIM * HDIM);
    int o   = idx % (EDIM * HDIM);
    int j   = o & 15;
    int l   = (o >> 4) & 31;
    int nt  = (o >> 9) & 3;
    int kt  = o >> 11;
    int n   = l & 15;
    int K   = ((l >> 4) << 4) + j;
    const float* W = (mat == 0) ? Wq : ((mat == 1) ? Wk : Wv);
    out[idx] = (__bf16)W[(kt * 32 + K) * HDIM + nt * 16 + n];
}

// ---------------- Kernel 1: fused QKV projection (x @ W + b), bf16 WMMA -----
// One wave per 16 rows of x; 12 accumulators = 3 matrices x 4 N-tiles.
// All 4 waves of a block consume IDENTICAL weight tiles, so the 12KB weight
// chunk of each K-step is staged into LDS by one TDM issue (wave 0), double
// buffered one step ahead: DMA latency overlaps a full K-step of WMMAs.
__global__ void __launch_bounds__(128, 1)
qkv_kernel(const float* __restrict__ x, const __bf16* __restrict__ wpk,
           const float* __restrict__ bq, const float* __restrict__ bk,
           const float* __restrict__ bv,
           __bf16* __restrict__ Qb, __bf16* __restrict__ Kb,
           __bf16* __restrict__ Vt) {
    __shared__ __align__(128) __bf16 wsh[2 * 6144];   // 2 x 12KB weight buffers

    const int lane = threadIdx.x & 31;
    const int wv   = threadIdx.x >> 5;
    const int wid  = blockIdx.x * (blockDim.x >> 5) + wv;
    const int gr   = wid * 16;              // global row base in [0, B*S)
    const int m    = lane & 15;
    const int half = lane >> 4;

    // Weight-chunk TDM descriptor: tile 2048 x 3 rows of 2-byte elems,
    // row stride E*H (one packed matrix); one issue moves Wq+Wk+Wv tiles.
    const u32x8 g1w = {0x00010000u,            // data_size = 2B
                       2048u << 16,            // tensor_dim0 = 2048
                       3u << 16,               // tensor_dim1 = 3
                       2048u << 16,            // tile_dim0 = 2048
                       3u,                     // tile_dim1 = 3
                       (unsigned)(EDIM * HDIM),// dim0 stride = 65536 elems
                       0u, 0u};

    v8f acc[12];
    v8f zz = {0.f,0.f,0.f,0.f,0.f,0.f,0.f,0.f};
#pragma unroll
    for (int i = 0; i < 12; ++i) acc[i] = zz;

    const float* xrow = x + (size_t)(gr + m) * EDIM;

    // Prologue: stage weights for kk=0 into buffer 0
    if (wv == 0) {
        tdm_load_2d((void*)wsh, 0u, (unsigned long long)(uintptr_t)wpk, g1w);
        asm volatile("s_wait_tensorcnt 0x0" ::: "memory");
    }
    __syncthreads();

#pragma unroll 1
    for (int kk = 0; kk < EDIM / 32; ++kk) {
        const int buf = kk & 1;
        const bool more = (kk + 1) < (EDIM / 32);

        // Kick off DMA of next K-step's weights into the other buffer
        if (more && wv == 0) {
            tdm_load_2d((void*)(wsh + (1 - buf) * 6144),
                        (unsigned)(1 - buf) * 12288u,
                        (unsigned long long)(uintptr_t)(wpk + (kk + 1) * 2048),
                        g1w);
        }

        if (more)                           // -> global_prefetch_b8
            __builtin_prefetch(xrow + (kk + 1) * 32, 0, 0);

        // A tile: 16x32 bf16, contiguous k = half*16 + j per lane
        const float* p0 = xrow + kk * 32 + half * 16;
        const float* p1 = p0 + 8;
        float4 f0 = *(const float4*)p0;
        float4 f1 = *(const float4*)(p0 + 4);
        float4 g0 = *(const float4*)p1;
        float4 g1 = *(const float4*)(p1 + 4);
        v16bf a;
        a[0]=(__bf16)f0.x; a[1]=(__bf16)f0.y; a[2]=(__bf16)f0.z; a[3]=(__bf16)f0.w;
        a[4]=(__bf16)f1.x; a[5]=(__bf16)f1.y; a[6]=(__bf16)f1.z; a[7]=(__bf16)f1.w;
        a[8]=(__bf16)g0.x; a[9]=(__bf16)g0.y; a[10]=(__bf16)g0.z; a[11]=(__bf16)g0.w;
        a[12]=(__bf16)g1.x; a[13]=(__bf16)g1.y; a[14]=(__bf16)g1.z; a[15]=(__bf16)g1.w;

        // B tiles from LDS (ds_load_b128 pairs, short latency)
        const __bf16* wb = wsh + buf * 6144 + lane * 16;
#pragma unroll
        for (int t = 0; t < 12; ++t) {
            int mat = t >> 2, nt = t & 3;
            v16bf bm = ldg_v16bf(wb + mat * 2048 + nt * 512);
            acc[t] = __builtin_amdgcn_wmma_f32_16x16x32_bf16(
                false, a, false, bm, (short)0, acc[t], false, false);
        }

        // Next buffer must be fully written (wave 0 waits the DMA) and all
        // waves done reading this one before it is overwritten next+1 step.
        if (more) {
            if (wv == 0)
                asm volatile("s_wait_tensorcnt 0x0" ::: "memory");
            __syncthreads();
        }
    }

    const int bIdx = gr / SDIM;
    const int sloc = gr - bIdx * SDIM;
#pragma unroll
    for (int nt = 0; nt < 4; ++nt) {
        int col = nt * 16 + m;
        float bqv = bq[col], bkv = bk[col], bvv = bv[col];
#pragma unroll
        for (int r = 0; r < 8; ++r) {
            int row = gr + half * 8 + r;
            Qb[(size_t)row * HDIM + col] = (__bf16)(acc[nt][r] + bqv);
            Kb[(size_t)row * HDIM + col] = (__bf16)(acc[4 + nt][r] + bkv);
            int s = sloc + half * 8 + r;
            Vt[(size_t)bIdx * HDIM * SDIM + (size_t)col * SDIM + s] =
                (__bf16)(acc[8 + nt][r] + bvv);
        }
    }
}

// ---------------- Kernel 2: causal flash attention, bf16 WMMA + TDM ---------
// One wave per 16-query tile; 32 keys/iteration. K (32x64) and V (64x32) bf16
// tiles are DMA'd into LDS by the Tensor Data Mover one iteration ahead.
__global__ void __launch_bounds__(128, 1)
attn_kernel(const __bf16* __restrict__ Qb, const __bf16* __restrict__ Kb,
            const __bf16* __restrict__ Vt, float* __restrict__ out) {
    // [0,2048): P-transpose buffers (512 elems/wave)
    // [2048 + wv*4096, +2048): K tile [32][64]; next 2048: V tile [64][32]
    __shared__ __align__(128) __bf16 shmem[2048 + 4 * 4096];

    const int lane  = threadIdx.x & 31;
    const int wv    = threadIdx.x >> 5;
    const int u     = blockIdx.x * 4 + wv;  // global q-tile id in [0, B*S/16)
    const int b     = u / (SDIM / 16);
    const int qt    = u % (SDIM / 16);
    const int qbase = qt * 16;
    const int m     = lane & 15;
    const int half  = lane >> 4;
    const int mbase = half * 8;

    __bf16* myl = shmem + wv * 512;
    __bf16* myK = shmem + 2048 + wv * 4096;
    __bf16* myV = myK + 2048;
    const unsigned kldsB = 4096u + (unsigned)wv * 8192u;  // byte offsets
    const unsigned vldsB = kldsB + 4096u;

    // TDM descriptor group1 constants (2-byte elems, 2D):
    // K: tensor 64 x 32 rows, tile 64x32, row stride 64
    const u32x8 g1k = {0x00010000u, 64u << 16, 32u << 16, 64u << 16,
                       32u, 64u, 0u, 0u};
    // V: tensor 32 x 64 rows, tile 32x64, row stride 4096 (Vt is [H][S])
    const u32x8 g1v = {0x00010000u, 32u << 16, 64u << 16, 32u << 16,
                       64u, 4096u, 0u, 0u};

    // Q A-tiles (16x32 bf16 layout), loaded once
    const __bf16* qrow = Qb + (size_t)(u * 16 + m) * HDIM;
    v16bf aq0 = ld2_v16bf(qrow + half * 16,      qrow + half * 16 + 8);
    v16bf aq1 = ld2_v16bf(qrow + 32 + half * 16, qrow + 32 + half * 16 + 8);

    v8f o0 = {0.f,0.f,0.f,0.f,0.f,0.f,0.f,0.f};
    v8f o1 = o0, o2 = o0, o3 = o0;
    float mrun[8], lrun[8];
#pragma unroll
    for (int r = 0; r < 8; ++r) { mrun[r] = -3.0e38f; lrun[r] = 0.0f; }

    const __bf16* Kbase = Kb + (size_t)b * SDIM * HDIM;
    const __bf16* Vbase = Vt + (size_t)b * HDIM * SDIM;
    const float sc = 0.125f;                // 1/sqrt(H)
    const int nit = (qbase + 47) / 32;      // cover keys 0 .. qbase+15

    // Prologue: DMA first K/V tile
    tdm_load_2d((void*)myK, kldsB, (unsigned long long)(uintptr_t)Kbase, g1k);
    tdm_load_2d((void*)myV, vldsB, (unsigned long long)(uintptr_t)Vbase, g1v);

#pragma unroll 1
    for (int it = 0; it < nit; ++it) {
        const int kb = it * 32;
        const bool maskit = (kb + 31) > qbase;

        asm volatile("s_wait_tensorcnt 0x0" ::: "memory");

        // LDS -> VGPR: K B-tiles (lane n = key, 16 contiguous h)
        v16bf bk00 = ldg_v16bf(myK + m * 64 + half * 16);
        v16bf bk01 = ldg_v16bf(myK + m * 64 + 32 + half * 16);
        v16bf bk10 = ldg_v16bf(myK + (16 + m) * 64 + half * 16);
        v16bf bk11 = ldg_v16bf(myK + (16 + m) * 64 + 32 + half * 16);
        // LDS -> VGPR: V B-tiles (lane n = h, 16 contiguous keys)
        v16bf bv0 = ldg_v16bf(myV + (0  + m) * 32 + half * 16);
        v16bf bv1 = ldg_v16bf(myV + (16 + m) * 32 + half * 16);
        v16bf bv2 = ldg_v16bf(myV + (32 + m) * 32 + half * 16);
        v16bf bv3 = ldg_v16bf(myV + (48 + m) * 32 + half * 16);

        asm volatile("s_wait_dscnt 0x0" ::: "memory");

        // DMA next tile while we compute on this one
        if (it + 1 < nit) {
            const int kb2 = kb + 32;
            tdm_load_2d((void*)myK, kldsB,
                (unsigned long long)(uintptr_t)(Kbase + (size_t)kb2 * HDIM), g1k);
            tdm_load_2d((void*)myV, vldsB,
                (unsigned long long)(uintptr_t)(Vbase + kb2), g1v);
        }

        v8f z = {0.f,0.f,0.f,0.f,0.f,0.f,0.f,0.f};
        v8f s0 = __builtin_amdgcn_wmma_f32_16x16x32_bf16(false, aq0, false, bk00, (short)0, z,  false, false);
        s0     = __builtin_amdgcn_wmma_f32_16x16x32_bf16(false, aq1, false, bk01, (short)0, s0, false, false);
        v8f s1 = __builtin_amdgcn_wmma_f32_16x16x32_bf16(false, aq0, false, bk10, (short)0, z,  false, false);
        s1     = __builtin_amdgcn_wmma_f32_16x16x32_bf16(false, aq1, false, bk11, (short)0, s1, false, false);

        // online softmax (rows live across 16-lane groups; width-16 shuffles)
#pragma unroll
        for (int r = 0; r < 8; ++r) {
            int q = qbase + mbase + r;
            float x0 = s0[r] * sc, x1 = s1[r] * sc;
            if (maskit) {
                if (kb + m > q)      x0 = -3.0e38f;
                if (kb + 16 + m > q) x1 = -3.0e38f;
            }
            float mx = fmaxf(x0, x1);
            mx = fmaxf(mx, __shfl_xor(mx, 1, 16));
            mx = fmaxf(mx, __shfl_xor(mx, 2, 16));
            mx = fmaxf(mx, __shfl_xor(mx, 4, 16));
            mx = fmaxf(mx, __shfl_xor(mx, 8, 16));
            float mn = fmaxf(mrun[r], mx);
            float al = __expf(mrun[r] - mn);
            float e0 = __expf(x0 - mn);
            float e1 = __expf(x1 - mn);
            float rs = e0 + e1;
            rs += __shfl_xor(rs, 1, 16);
            rs += __shfl_xor(rs, 2, 16);
            rs += __shfl_xor(rs, 4, 16);
            rs += __shfl_xor(rs, 8, 16);
            lrun[r] = lrun[r] * al + rs;
            mrun[r] = mn;
            o0[r] *= al; o1[r] *= al; o2[r] *= al; o3[r] *= al;
            // transpose P (C layout -> A layout) through LDS, as bf16
            myl[(mbase + r) * 32 + m]      = (__bf16)e0;
            myl[(mbase + r) * 32 + 16 + m] = (__bf16)e1;
        }
        __builtin_amdgcn_wave_barrier();    // DS ops are wave-in-order

        v16bf pa = ld2_v16bf(myl + m * 32 + half * 16,
                             myl + m * 32 + half * 16 + 8);

        o0 = __builtin_amdgcn_wmma_f32_16x16x32_bf16(false, pa, false, bv0, (short)0, o0, false, false);
        o1 = __builtin_amdgcn_wmma_f32_16x16x32_bf16(false, pa, false, bv1, (short)0, o1, false, false);
        o2 = __builtin_amdgcn_wmma_f32_16x16x32_bf16(false, pa, false, bv2, (short)0, o2, false, false);
        o3 = __builtin_amdgcn_wmma_f32_16x16x32_bf16(false, pa, false, bv3, (short)0, o3, false, false);
    }

#pragma unroll
    for (int r = 0; r < 8; ++r) {
        float inv = 1.0f / lrun[r];
        size_t row = (size_t)(u * 16 + mbase + r) * HDIM;
        out[row +  0 + m] = o0[r] * inv;
        out[row + 16 + m] = o1[r] * inv;
        out[row + 32 + m] = o2[r] * inv;
        out[row + 48 + m] = o3[r] * inv;
    }
}

extern "C" void kernel_launch(void* const* d_in, const int* in_sizes, int n_in,
                              void* d_out, int out_size, void* d_ws, size_t ws_size,
                              hipStream_t stream) {
    (void)in_sizes; (void)n_in; (void)out_size; (void)ws_size;
    const float* x  = (const float*)d_in[0];
    const float* Wq = (const float*)d_in[1];
    const float* bq = (const float*)d_in[2];
    const float* Wk = (const float*)d_in[3];
    const float* bk = (const float*)d_in[4];
    const float* Wv = (const float*)d_in[5];
    const float* bv = (const float*)d_in[6];
    float* out = (float*)d_out;

    char* ws = (char*)d_ws;
    const size_t QKV_BYTES = (size_t)BDIM * SDIM * HDIM * 2; // 2 MB each
    __bf16* wpk = (__bf16*)ws;                               // 384 KB used
    __bf16* Qb  = (__bf16*)(ws + 524288);
    __bf16* Kb  = (__bf16*)(ws + 524288 + QKV_BYTES);
    __bf16* Vt  = (__bf16*)(ws + 524288 + 2 * QKV_BYTES);

    // Kernel 0: pack weights into WMMA B-tile bf16 layout
    pack_w_kernel<<<(3 * EDIM * HDIM + 255) / 256, 256, 0, stream>>>(Wq, Wk, Wv, wpk);

    // Kernel 1: fused QKV projection (1024 waves, 4 waves/block),
    // weights staged block-wide via TDM double buffering
    qkv_kernel<<<(BDIM * SDIM / 16) / 4, 128, 0, stream>>>(x, wpk, bq, bk, bv, Qb, Kb, Vt);

    // Kernel 2: causal flash attention with TDM staging (1024 waves)
    attn_kernel<<<(BDIM * SDIM / 16) / 4, 128, 0, stream>>>(Qb, Kb, Vt, out);
}
